// DynSMHALayer_82411832475901
// MI455X (gfx1250) — matrix operand order repaired and checked by reading, not compiled, not run on hardware
//
#include <hip/hip_runtime.h>
#include <hip/hip_bf16.h>
#include <stdint.h>

// ---------------------------------------------------------------------------
// DynSMHA layer for MI455X (gfx1250): bf16 WMMA GEMMs + flash attention,
// with Tensor Data Mover (TDM) async global->LDS tile loads.
// B=8, T=2048, C=1024, E=16, D=64, N=16384.
// ---------------------------------------------------------------------------

#define N_TOK 16384
#define CDIM  1024
#define EXP   16
#define DDIM  64
#define BATCH 8
#define TSEQ  2048

typedef __attribute__((ext_vector_type(16))) __bf16 v16bf;
typedef __attribute__((ext_vector_type(8)))  float  v8f;
typedef unsigned int u32x4 __attribute__((ext_vector_type(4)));
typedef int          i32x4 __attribute__((ext_vector_type(4)));
typedef int          i32x8 __attribute__((ext_vector_type(8)));

union FragBF {
    v16bf v;
    uint4 q[2];
    unsigned short h[16];
};

__device__ __forceinline__ unsigned short f2bf(float f) {
    unsigned u = __float_as_uint(f);
    u += 0x7FFFu + ((u >> 16) & 1u);      // round-to-nearest-even
    return (unsigned short)(u >> 16);
}
__device__ __forceinline__ float bf2f(unsigned short h) {
    return __uint_as_float(((unsigned)h) << 16);
}

// ---- TDM: async 2D tile load global->LDS (D# per CDNA5 ISA ch.8) -----------
// tile_d0 elems per row (bf16), tile_d1 rows; tensor row stride = stride elems.
// pad_i / pad_a encode LDS padding: interval = 8<<pad_i bytes, pad = (pad_a+1) DW.
__device__ __forceinline__ void tdm_load_2d(unsigned lds_off, const void* gp,
                                            unsigned tile_d0, unsigned tile_d1,
                                            unsigned tensor_d0, unsigned tensor_d1,
                                            unsigned stride_e,
                                            unsigned pad_i, unsigned pad_a) {
    unsigned long long ga = (unsigned long long)(uintptr_t)gp;
    u32x4 g0;
    g0[0] = 1u;                                            // count=1, user mode
    g0[1] = lds_off;                                       // LDS byte address
    g0[2] = (unsigned)(ga & 0xFFFFFFFFu);                  // global_addr lo
    g0[3] = (unsigned)((ga >> 32) & 0x1FFFFFFu) | 0x80000000u;  // hi | type=2
    i32x8 g1;
    g1[0] = (int)((1u << 16) |                 // data_size = 2 bytes
                  (1u << 20) |                 // pad_enable
                  (pad_i << 22) | (pad_a << 25));
    g1[1] = (int)(tensor_d0 << 16);                        // tensor_dim0 lo16
    g1[2] = (int)((tensor_d0 >> 16) | (tensor_d1 << 16));  // dim0 hi / dim1 lo
    g1[3] = (int)((tensor_d1 >> 16) | (tile_d0 << 16));    // dim1 hi / tile0
    g1[4] = (int)(tile_d1 & 0xFFFF);                       // tile1 (tile2=0)
    g1[5] = (int)stride_e;                                 // dim0_stride lo32
    g1[6] = 0;
    g1[7] = 0;
    i32x4 z4 = {0, 0, 0, 0};
#if __clang_major__ >= 23
    i32x8 z8 = {0, 0, 0, 0, 0, 0, 0, 0};
    __builtin_amdgcn_tensor_load_to_lds(g0, g1, z4, z4, z8, 0);
#else
    __builtin_amdgcn_tensor_load_to_lds(g0, g1, z4, z4, 0);
#endif
}

// ---------------- prep kernels ----------------

__global__ void k_f32_to_bf16(const float* __restrict__ in,
                              unsigned short* __restrict__ out, int n) {
    int i = blockIdx.x * 256 + threadIdx.x;
    if (i < n) out[i] = f2bf(in[i]);
}

// Wqkv[c][s*1024 + e*64 + d] = proj_s[e][c][d],  shape [1024, 3072] bf16
__global__ void k_pack_wqkv(const float* __restrict__ q_,
                            const float* __restrict__ k_,
                            const float* __restrict__ v_,
                            unsigned short* __restrict__ w) {
    int i = blockIdx.x * 256 + threadIdx.x;   // [0, 1024*3072)
    int c = i / 3072;
    int r = i - c * 3072;
    int s = r >> 10;
    int rq = r & 1023;
    int e = rq >> 6;
    int d = rq & 63;
    const float* src = (s == 0) ? q_ : (s == 1) ? k_ : v_;
    w[i] = f2bf(src[(size_t)e * CDIM * DDIM + (size_t)c * DDIM + d]);
}

// column-normalize sim_matrix [C,E] -> snN [C,E] (same layout)
__global__ void k_sim_norm(const float* __restrict__ sim, float* __restrict__ snN) {
    __shared__ float red[256];
    int e = blockIdx.x, tid = threadIdx.x;
    float ss = 0.f;
    for (int c = tid; c < CDIM; c += 256) { float s = sim[c * EXP + e]; ss += s * s; }
    red[tid] = ss; __syncthreads();
    for (int o = 128; o > 0; o >>= 1) { if (tid < o) red[tid] += red[tid + o]; __syncthreads(); }
    float inv = 1.0f / fmaxf(sqrtf(red[0]), 1e-12f);
    for (int c = tid; c < CDIM; c += 256) snN[c * EXP + e] = sim[c * EXP + e] * inv;
}

// ---------------- gating: 16 lanes (one half-wave) = one token ----------------
__global__ void __launch_bounds__(256) k_gating(const float* __restrict__ X,
                                                const float* __restrict__ snN,
                                                const float* __restrict__ gates,
                                                float* __restrict__ rw) {
    int tid = threadIdx.x, lane = tid & 31;
    int half = lane >> 4, e = lane & 15;
    int t = blockIdx.x * 16 + (tid >> 5) * 2 + half;
    const float* x = X + (size_t)t * CDIM;
    const float* sn = snN + e;
    float acc = 0.f, ss = 0.f;
    for (int c = 0; c < CDIM; c += 4) {
#pragma unroll
        for (int u = 0; u < 4; ++u) {
            float xi = x[c + u];
            acc = fmaf(xi, sn[(c + u) * EXP], acc);
            ss  = fmaf(xi, xi, ss);
        }
    }
    float logit = acc / fmaxf(sqrtf(ss), 1e-12f) - gates[e];
    float gated = fmaxf(logit, 0.f);
    bool act = gated > 0.f;
    unsigned long long bal = __ballot(act);
    bool inactive = (((bal >> (half * 16)) & 0xFFFFull) == 0ull);
    // top-2 of raw logits within the 16-lane group (lower index wins ties)
    float m1 = logit; int i1 = e;
#pragma unroll
    for (int o = 8; o; o >>= 1) {
        float om = __shfl_xor(m1, o, 32); int oi = __shfl_xor(i1, o, 32);
        if (om > m1 || (om == m1 && oi < i1)) { m1 = om; i1 = oi; }
    }
    float m2 = (e == i1) ? -3.0e38f : logit; int i2 = (e == i1) ? 99 : e;
#pragma unroll
    for (int o = 8; o; o >>= 1) {
        float om = __shfl_xor(m2, o, 32); int oi = __shfl_xor(i2, o, 32);
        if (om > m2 || (om == m2 && oi < i2)) { m2 = om; i2 = oi; }
    }
    bool fin = inactive ? (e == i1 || e == i2) : act;
    float mval = fin ? gated : -3.0e38f;
    float mx = mval;
#pragma unroll
    for (int o = 8; o; o >>= 1) mx = fmaxf(mx, __shfl_xor(mx, o, 32));
    float p = fin ? __expf(gated - mx) : 0.f;
    float sum = p;
#pragma unroll
    for (int o = 8; o; o >>= 1) sum += __shfl_xor(sum, o, 32);
    rw[(size_t)t * EXP + e] = p / sum;
}

// ---------------- tiled bf16 WMMA GEMM: C[M,N] = A[M,K] * B[K,N] ----------------
// Block tile 128x64, BK=32, 256 threads = 8 waves, wave -> 32x32 of output.
// A tile moved by the Tensor Data Mover (async, TENSORcnt); B tile transposed
// into LDS by the wave lanes.
__global__ void __launch_bounds__(256) k_gemm_bf16(const unsigned short* __restrict__ A,
                                                   const unsigned short* __restrict__ B,
                                                   void* __restrict__ Cout,
                                                   int M, int Ncols, int K, int out_bf16) {
    __shared__ unsigned short lsA[128 * 40];  // [row][k], stride 40 (TDM pad)
    __shared__ unsigned short lsBT[64 * 40];  // [n][k] transposed, stride 40
    int tid = threadIdx.x, lane = tid & 31, wid = tid >> 5;
    int bm = blockIdx.x, bn = blockIdx.y;
    const size_t ldb = (size_t)Ncols;
    int mrow = (wid & 3) * 32;      // wave M base within block (0..96)
    int nb   = (wid >> 2) * 32;     // wave N base within block (0 or 32)
    v8f acc[2][2];
#pragma unroll
    for (int i = 0; i < 2; ++i)
#pragma unroll
        for (int jc = 0; jc < 2; ++jc)
#pragma unroll
            for (int j = 0; j < 8; ++j) acc[i][jc][j] = 0.f;
    int brow = tid >> 3, bcol = (tid & 7) << 3;    // B tile: 32 k x 64 n
    const unsigned short* Bp = B + (size_t)brow * ldb + (size_t)bn * 64 + bcol;
    const unsigned short* Atile = A + (size_t)(bm * 128) * (size_t)K;
    unsigned lsA_off = (unsigned)(unsigned long long)(uintptr_t)&lsA[0];
    int frow = lane & 15, half = lane >> 4;
    for (int kb = 0; kb < K; kb += 32) {
        // async TDM load of the 128x32 A tile (64B rows + 16B LDS pad -> stride 80B)
        if (wid == 0)
            tdm_load_2d(lsA_off, Atile + kb, 32u, 128u,
                        (unsigned)K, (unsigned)M, (unsigned)K, 3u, 3u);
        // B tile: load 8 bf16 per lane, store transposed
        uint4 bvv = *(const uint4*)(Bp + (size_t)kb * ldb);
        const unsigned short* bh = (const unsigned short*)&bvv;
#pragma unroll
        for (int i = 0; i < 8; ++i) lsBT[(bcol + i) * 40 + brow] = bh[i];
        if (wid == 0) __builtin_amdgcn_s_wait_tensorcnt(0);
        __syncthreads();
        FragBF a0, a1, b0, b1;
        int ar0 = (mrow + frow) * 40 + half * 8;
        a0.q[0] = *(const uint4*)&lsA[ar0];
        a0.q[1] = *(const uint4*)&lsA[ar0 + 16];
        int ar1 = (mrow + 16 + frow) * 40 + half * 8;
        a1.q[0] = *(const uint4*)&lsA[ar1];
        a1.q[1] = *(const uint4*)&lsA[ar1 + 16];
        int br0 = (nb + frow) * 40 + half * 8;
        b0.q[0] = *(const uint4*)&lsBT[br0];
        b0.q[1] = *(const uint4*)&lsBT[br0 + 16];
        int br1 = (nb + 16 + frow) * 40 + half * 8;
        b1.q[0] = *(const uint4*)&lsBT[br1];
        b1.q[1] = *(const uint4*)&lsBT[br1 + 16];
        acc[0][0] = __builtin_amdgcn_wmma_f32_16x16x32_bf16(false, a0.v, false, b0.v,
                                                            (short)0, acc[0][0], false, false);
        acc[0][1] = __builtin_amdgcn_wmma_f32_16x16x32_bf16(false, a0.v, false, b1.v,
                                                            (short)0, acc[0][1], false, false);
        acc[1][0] = __builtin_amdgcn_wmma_f32_16x16x32_bf16(false, a1.v, false, b0.v,
                                                            (short)0, acc[1][0], false, false);
        acc[1][1] = __builtin_amdgcn_wmma_f32_16x16x32_bf16(false, a1.v, false, b1.v,
                                                            (short)0, acc[1][1], false, false);
        __syncthreads();
    }
#pragma unroll
    for (int i = 0; i < 2; ++i) {
        int orow0 = bm * 128 + mrow + i * 16 + half * 8;
#pragma unroll
        for (int jc = 0; jc < 2; ++jc) {
            int ocol = bn * 64 + nb + jc * 16 + frow;
            if (out_bf16) {
                unsigned short* O = (unsigned short*)Cout;
#pragma unroll
                for (int j = 0; j < 8; ++j)
                    O[(size_t)(orow0 + j) * ldb + ocol] = f2bf(acc[i][jc][j]);
            } else {
                float* O = (float*)Cout;
#pragma unroll
                for (int j = 0; j < 8; ++j)
                    O[(size_t)(orow0 + j) * ldb + ocol] = acc[i][jc][j];
            }
        }
    }
}

// ---------------- combine experts -> q,k,v (bf16); softmax scale folded into q ----
__global__ void k_combine_qkv(const unsigned short* __restrict__ Xqkv,
                              const float* __restrict__ rw,
                              unsigned short* __restrict__ qb,
                              unsigned short* __restrict__ kb,
                              unsigned short* __restrict__ vb) {
    int tid = threadIdx.x;
    int t = blockIdx.x * 4 + (tid >> 6);
    int d = tid & 63;
    const unsigned short* row = Xqkv + (size_t)t * 3072;
    const float* r = rw + (size_t)t * EXP;
    float q = 0.f, k = 0.f, v = 0.f;
#pragma unroll
    for (int e = 0; e < EXP; ++e) {
        float w = r[e];
        q = fmaf(w, bf2f(row[e * 64 + d]), q);
        k = fmaf(w, bf2f(row[1024 + e * 64 + d]), k);
        v = fmaf(w, bf2f(row[2048 + e * 64 + d]), v);
    }
    qb[(size_t)t * 64 + d] = f2bf(q * 0.125f);   // 1/sqrt(64)
    kb[(size_t)t * 64 + d] = f2bf(k);
    vb[(size_t)t * 64 + d] = f2bf(v);
}

// ---------------- flash attention: one head per batch, 128 q-rows per block ------
// K tile moved by TDM; V tile transposed into LDS by the lanes.
__global__ void __launch_bounds__(256) k_attn(const unsigned short* __restrict__ qb,
                                              const unsigned short* __restrict__ kbp,
                                              const unsigned short* __restrict__ vbp,
                                              float* __restrict__ ao) {
    __shared__ unsigned short lsK[32 * 72];        // [key][d], stride 72 (TDM pad)
    __shared__ unsigned short lsVT[64 * 40];       // [d][key] transposed
    __shared__ unsigned short lsP[8 * 16 * 40];    // per-wave P relayout scratch
    int tid = threadIdx.x, lane = tid & 31, wid = tid >> 5;
    size_t base = (size_t)blockIdx.y * TSEQ * 64;
    int qrow0 = blockIdx.x * 128 + wid * 16;
    int frow = lane & 15, half = lane >> 4;
    FragBF aq[2];
    {
        const unsigned short* qp = qb + base + (size_t)(qrow0 + frow) * 64 + half * 8;
        aq[0].q[0] = *(const uint4*)(qp);
        aq[0].q[1] = *(const uint4*)(qp + 16);
        aq[1].q[0] = *(const uint4*)(qp + 32);
        aq[1].q[1] = *(const uint4*)(qp + 48);
    }
    float m[8], l[8];
    v8f O[4];
#pragma unroll
    for (int j = 0; j < 8; ++j) {
        m[j] = -3.0e38f; l[j] = 0.f;
        O[0][j] = O[1][j] = O[2][j] = O[3][j] = 0.f;
    }
    int ldr = tid >> 3, ldc = (tid & 7) << 3;
    unsigned short* myP = lsP + wid * 640;
    unsigned lsK_off = (unsigned)(unsigned long long)(uintptr_t)&lsK[0];
    for (int kb0 = 0; kb0 < TSEQ; kb0 += 32) {
        // async TDM load of the 32x64 K tile (128B rows + 16B pad -> stride 144B)
        if (wid == 0)
            tdm_load_2d(lsK_off, kbp + base + (size_t)kb0 * 64, 64u, 32u,
                        64u, (unsigned)(BATCH * TSEQ), 64u, 4u, 3u);
        uint4 vv = *(const uint4*)(vbp + base + (size_t)(kb0 + ldr) * 64 + ldc);
        const unsigned short* vh = (const unsigned short*)&vv;
#pragma unroll
        for (int i = 0; i < 8; ++i) lsVT[(ldc + i) * 40 + ldr] = vh[i];
        if (wid == 0) __builtin_amdgcn_s_wait_tensorcnt(0);
        __syncthreads();
        // S = q @ k^T over d=64 (two K=32 steps), two 16-key tiles
        v8f S0, S1;
#pragma unroll
        for (int j = 0; j < 8; ++j) { S0[j] = 0.f; S1[j] = 0.f; }
#pragma unroll
        for (int g = 0; g < 2; ++g) {
            FragBF bk0, bk1;
            int off = g * 32 + half * 8;
            const unsigned short* p0 = &lsK[frow * 72 + off];
            bk0.q[0] = *(const uint4*)p0; bk0.q[1] = *(const uint4*)(p0 + 16);
            const unsigned short* p1 = &lsK[(16 + frow) * 72 + off];
            bk1.q[0] = *(const uint4*)p1; bk1.q[1] = *(const uint4*)(p1 + 16);
            S0 = __builtin_amdgcn_wmma_f32_16x16x32_bf16(false, aq[g].v, false, bk0.v,
                                                         (short)0, S0, false, false);
            S1 = __builtin_amdgcn_wmma_f32_16x16x32_bf16(false, aq[g].v, false, bk1.v,
                                                         (short)0, S1, false, false);
        }
        // online softmax (row reductions across 16 lanes of each half)
#pragma unroll
        for (int j = 0; j < 8; ++j) {
            float loc = fmaxf(S0[j], S1[j]);
#pragma unroll
            for (int o = 8; o; o >>= 1) loc = fmaxf(loc, __shfl_xor(loc, o, 32));
            float mn = fmaxf(m[j], loc);
            float alpha = __expf(m[j] - mn);
            m[j] = mn;
            float p0 = __expf(S0[j] - mn);
            float p1 = __expf(S1[j] - mn);
            float rs = p0 + p1;
#pragma unroll
            for (int o = 8; o; o >>= 1) rs += __shfl_xor(rs, o, 32);
            l[j] = l[j] * alpha + rs;
            O[0][j] *= alpha; O[1][j] *= alpha; O[2][j] *= alpha; O[3][j] *= alpha;
            int pr = (j + half * 8) * 40;
            myP[pr + frow]      = f2bf(p0);
            myP[pr + 16 + frow] = f2bf(p1);
        }
        // wave-local LDS roundtrip to re-layout P (C-layout) into A-fragment layout
        asm volatile("s_wait_dscnt 0x0" ::: "memory");
        FragBF ap;
        int apo = frow * 40 + half * 8;
        ap.q[0] = *(const uint4*)&myP[apo];
        ap.q[1] = *(const uint4*)&myP[apo + 16];
        // O += P @ V (K=32 keys, 4 d-tiles)
#pragma unroll
        for (int n = 0; n < 4; ++n) {
            FragBF bv2;
            const unsigned short* pv = &lsVT[(n * 16 + frow) * 40 + half * 8];
            bv2.q[0] = *(const uint4*)pv;
            bv2.q[1] = *(const uint4*)(pv + 16);
            O[n] = __builtin_amdgcn_wmma_f32_16x16x32_bf16(false, ap.v, false, bv2.v,
                                                           (short)0, O[n], false, false);
        }
        __syncthreads();
    }
#pragma unroll
    for (int j = 0; j < 8; ++j) {
        float inv = 1.0f / l[j];
        int row = qrow0 + j + half * 8;
#pragma unroll
        for (int n = 0; n < 4; ++n)
            ao[base + (size_t)row * 64 + n * 16 + frow] = O[n][j] * inv;
    }
}

// ---------------- u[t, e*64+d] = rw[t,e] * attn_out[t,d]  (bf16) ----------------
__global__ void k_build_u(const float* __restrict__ ao, const float* __restrict__ rw,
                          unsigned short* __restrict__ u) {
    int i = blockIdx.x * 256 + threadIdx.x;   // N*1024
    int t = i >> 10, col = i & 1023, e = col >> 6, d = col & 63;
    u[i] = f2bf(rw[(size_t)t * EXP + e] * ao[(size_t)t * 64 + d]);
}

// ---------------------------------------------------------------------------

extern "C" void kernel_launch(void* const* d_in, const int* in_sizes, int n_in,
                              void* d_out, int out_size, void* d_ws, size_t ws_size,
                              hipStream_t stream) {
    (void)in_sizes; (void)n_in; (void)out_size; (void)ws_size;
    const float* hs    = (const float*)d_in[0];
    const float* sim   = (const float*)d_in[1];
    const float* gates = (const float*)d_in[2];
    const float* qp    = (const float*)d_in[3];
    const float* kp    = (const float*)d_in[4];
    const float* vp    = (const float*)d_in[5];
    const float* op    = (const float*)d_in[6];
    float* out = (float*)d_out;
    char* ws = (char*)d_ws;

    size_t off = 0;
    auto alloc = [&](size_t bytes) {
        size_t o = off;
        off = (off + bytes + 255) & ~(size_t)255;
        return o;
    };
    unsigned short* Xbf  = (unsigned short*)(ws + alloc((size_t)N_TOK * CDIM * 2));
    unsigned short* Wqkv = (unsigned short*)(ws + alloc((size_t)CDIM * 3072 * 2));
    unsigned short* Wo   = (unsigned short*)(ws + alloc((size_t)CDIM * CDIM * 2));
    float*          snN  = (float*)(ws + alloc((size_t)CDIM * EXP * 4));
    float*          rw   = (float*)(ws + alloc((size_t)N_TOK * EXP * 4));
    unsigned short* Xqkv = (unsigned short*)(ws + alloc((size_t)N_TOK * 3072 * 2));
    unsigned short* qb   = (unsigned short*)(ws + alloc((size_t)N_TOK * DDIM * 2));
    unsigned short* kb   = (unsigned short*)(ws + alloc((size_t)N_TOK * DDIM * 2));
    unsigned short* vb   = (unsigned short*)(ws + alloc((size_t)N_TOK * DDIM * 2));
    float*          ao   = (float*)(ws + alloc((size_t)N_TOK * DDIM * 4));
    unsigned short* u    = (unsigned short*)(ws + alloc((size_t)N_TOK * CDIM * 2));

    // prep
    k_f32_to_bf16<<<(N_TOK * CDIM) / 256, 256, 0, stream>>>(hs, Xbf, N_TOK * CDIM);
    k_f32_to_bf16<<<(CDIM * CDIM) / 256, 256, 0, stream>>>(op, Wo, CDIM * CDIM);
    k_pack_wqkv<<<(CDIM * 3072) / 256, 256, 0, stream>>>(qp, kp, vp, Wqkv);
    k_sim_norm<<<EXP, 256, 0, stream>>>(sim, snN);
    // gating
    k_gating<<<N_TOK / 16, 256, 0, stream>>>(hs, snN, gates, rw);
    // per-expert QKV projections as one [16384,1024]x[1024,3072] bf16 WMMA GEMM
    k_gemm_bf16<<<dim3(N_TOK / 128, 3072 / 64), 256, 0, stream>>>(Xbf, Wqkv, Xqkv,
                                                                  N_TOK, 3072, CDIM, 1);
    k_combine_qkv<<<N_TOK / 4, 256, 0, stream>>>(Xqkv, rw, qb, kb, vb);
    // flash attention per batch
    k_attn<<<dim3(TSEQ / 128, BATCH), 256, 0, stream>>>(qb, kb, vb, ao);
    // output projection: u[16384,1024] @ Wo[1024,1024] -> f32 out
    k_build_u<<<(N_TOK * CDIM) / 256, 256, 0, stream>>>(ao, rw, u);
    k_gemm_bf16<<<dim3(N_TOK / 128, CDIM / 64), 256, 0, stream>>>(u, Wo, out,
                                                                  N_TOK, CDIM, CDIM, 0);
}
